// EfficientGSA_31817117729224
// MI455X (gfx1250) — matrix operand order repaired and checked by reading, not compiled, not run on hardware
//
#include <hip/hip_runtime.h>

// Problem constants (from reference): B=4, T=1024, D=512, H=8, S=16, HD=64
#define Bc 4
#define Tc 1024
#define Dc 512
#define Hc 8
#define Sc 16
#define HDc 64
#define MTOK (Bc * Tc)          // 4096 tokens
#define QKVN (3 * Dc)           // 1536

typedef __attribute__((ext_vector_type(16))) _Float16 v16h;
typedef __attribute__((ext_vector_type(8)))  _Float16 v8h;
typedef __attribute__((ext_vector_type(8)))  float    v8f;
typedef __attribute__((ext_vector_type(4)))  int      v4i;

#ifndef __has_builtin
#define __has_builtin(x) 0
#endif
#if __has_builtin(__builtin_amdgcn_global_load_async_to_lds_b128) && \
    __has_builtin(__builtin_amdgcn_s_wait_asynccnt)
#define HAVE_ASYNC_LDS 1
#else
#define HAVE_ASYNC_LDS 0
#endif

#define AS1 __attribute__((address_space(1)))
#define AS3 __attribute__((address_space(3)))

static __device__ __forceinline__ v8f wmma16(v16h a, v16h b, v8f c) {
  // D(f32 16x16) = A(f16 16x32) * B(f16 32x16) + C
  return __builtin_amdgcn_wmma_f32_16x16x32_f16(
      /*neg_a=*/false, a, /*neg_b=*/false, b,
      /*c_mod=*/(short)0, c, /*reuse_a=*/false, /*reuse_b=*/false);
}

// 16-byte global -> LDS copy, async on gfx1250 toolchains that expose it
static __device__ __forceinline__ void copy16_g2l(const _Float16* src,
                                                  _Float16* lds_dst) {
#if HAVE_ASYNC_LDS
  __builtin_amdgcn_global_load_async_to_lds_b128(
      (AS1 v4i*)(void*)src, (AS3 v4i*)(void*)lds_dst, 0, 0);
#else
  *(v8h*)lds_dst = *(const v8h*)src;
#endif
}
static __device__ __forceinline__ void wait_g2l() {
#if HAVE_ASYNC_LDS
  __builtin_amdgcn_s_wait_asynccnt(0);
#endif
}

// ---------------------------------------------------------------------------
// Kernel 1: splat parameter preprocessing (tiny). P layout (floats):
//   [0..8191]     centers[h][s][d]
//   [8192..8319]  c_sq[h*16+s]
//   [8320..8447]  inv_var
//   [8448..8575]  eff_amps
//   [8576]        inv_temp
__global__ void prep_params(const float* __restrict__ base_centers,
                            const float* __restrict__ center_deltas,
                            const float* __restrict__ log_scales,
                            const float* __restrict__ log_amps,
                            const float* __restrict__ msp,
                            const float* __restrict__ temperature,
                            float* __restrict__ P) {
  int j = threadIdx.x;                     // 0..127 == h*16+s
  float ms = 0.1f / (1.0f + __expf(-msp[0]));
  float sc = __expf(log_scales[j]);
  sc = fminf(fmaxf(sc, 1e-4f), 2.0f);
  float iv = 0.5f / (sc * sc + 1e-8f);
  float amp = __expf(log_amps[j]);
  float ea = (amp > 0.02f) ? amp : 0.0f;
  float csq = 0.0f;
  #pragma unroll 4
  for (int d = 0; d < HDc; ++d) {
    float c = base_centers[j * HDc + d] + center_deltas[j * HDc + d] * ms;
    P[j * HDc + d] = c;
    csq += c * c;
  }
  P[8192 + j] = csq;
  P[8320 + j] = iv;
  P[8448 + j] = ea;
  if (j == 0) P[8576] = 1.0f / fmaxf(temperature[0], 0.1f);
}

// ---------------------------------------------------------------------------
// Kernel 2: f32 -> f16 conversion (x, qkv_w, out_w)
__global__ void to_f16(const float* __restrict__ src, _Float16* __restrict__ dst,
                       size_t n) {
  size_t i = (size_t)blockIdx.x * blockDim.x + threadIdx.x;
  size_t stride = (size_t)gridDim.x * blockDim.x;
  for (; i < n; i += stride) dst[i] = (_Float16)src[i];
}

// ---------------------------------------------------------------------------
// Kernel 3: WMMA GEMM  C[M,N](f32) = A[M,K](f16) * B[N,K]^T(f16)
// Block (256 thr, 8 waves) computes a 128x64 C tile. The 64x32 B panel for
// each K-step is staged once in LDS via async global->LDS copies and shared
// by all 8 waves; A fragments stream from global. K % 32 == 0.
__global__ void __launch_bounds__(256)
gemm_f16_wmma(const _Float16* __restrict__ A, const _Float16* __restrict__ Bw,
              float* __restrict__ C, int M, int N, int K) {
  __shared__ _Float16 bsh[64 * 32];           // 4 KB B panel
  const int lane = threadIdx.x & 31;
  const int wv = threadIdx.x >> 5;            // 0..7
  const int nblkN = N >> 6;
  const int bm = blockIdx.x / nblkN;
  const int bn = blockIdx.x - bm * nblkN;
  const int row0 = bm * 128 + wv * 16;
  const int col0 = bn * 64;
  const int m = lane & 15, hi = lane >> 4;
  const int kbA = hi ? 16 : 0;    // A k order used here (contiguous): lo-half K=0..15
  const int kbB = hi ? 16 : 0;    // B k order used here (contiguous): lo-half K=0..15
  // cooperative copy indices: 64 rows x 32 halfs, 16B per thread
  const int crow = threadIdx.x >> 2;          // 0..63
  const int cseg = (threadIdx.x & 3) * 8;     // 0,8,16,24 halfs
  const bool valid = row0 < M;
  v8f acc[4] = {};
  const _Float16* arow = A + (size_t)(row0 + m) * K;

  for (int k0 = 0; k0 < K; k0 += 32) {
    // stage B panel: Bw[col0+crow][k0+cseg .. +7] -> bsh[crow*32+cseg]
    copy16_g2l(Bw + (size_t)(col0 + crow) * K + k0 + cseg,
               &bsh[crow * 32 + cseg]);
    wait_g2l();
    __syncthreads();
    if (valid) {
      v16h a;
      {
        const _Float16* p = arow + k0 + kbA;
        #pragma unroll
        for (int t = 0; t < 16; ++t) a[t] = p[t];
      }
      __builtin_prefetch(arow + k0 + 64, 0, 3);
      #pragma unroll
      for (int g = 0; g < 4; ++g) {
        v16h b;
        const _Float16* p = &bsh[(g * 16 + m) * 32 + kbB];
        #pragma unroll
        for (int t = 0; t < 16; ++t) b[t] = p[t];
        acc[g] = wmma16(a, b, acc[g]);
      }
    }
    __syncthreads();
  }
  if (!valid) return;
  // C layout: VGPR r -> row (r + 8*hi), col = lane&15 within 16-col group
  #pragma unroll
  for (int g = 0; g < 4; ++g)
    #pragma unroll
    for (int r = 0; r < 8; ++r)
      C[(size_t)(row0 + r + hi * 8) * N + (col0 + g * 16 + m)] = acc[g][r];
}

// ---------------------------------------------------------------------------
// Kernel 4: splat weights qw,kw[tok][h][s] (f16) from q,k rows of qkv (f32)
__global__ void __launch_bounds__(128)
splat_weights(const float* __restrict__ qkv, const float* __restrict__ P,
              _Float16* __restrict__ qw, _Float16* __restrict__ kw) {
  const int tok = blockIdx.x;          // 0..4095
  const int j = threadIdx.x;           // 0..127 == h*16+s
  const int h = j >> 4;
  const float csq = P[8192 + j], iv = P[8320 + j], ea = P[8448 + j];
  const float* q = qkv + (size_t)tok * QKVN + h * HDc;        // cols [0,512)
  const float* k = q + Dc;                                    // cols [512,1024)
  const float* c = P + j * HDc;
  float dq = 0.f, qq = 0.f, dk = 0.f, kk = 0.f;
  #pragma unroll 4
  for (int d = 0; d < HDc; ++d) {
    float qv = q[d], kv = k[d], cv = c[d];
    dq += qv * cv; qq += qv * qv;
    dk += kv * cv; kk += kv * kv;
  }
  float d2q = qq + csq - 2.0f * dq;
  float d2k = kk + csq - 2.0f * dk;
  qw[(size_t)tok * 128 + j] = (_Float16)(__expf(-d2q * iv) * ea);
  kw[(size_t)tok * 128 + j] = (_Float16)(__expf(-d2k * iv) * ea);
}

// ---------------------------------------------------------------------------
// Kernel 5: transpose V to f16 [b][h][d][t] for contiguous WMMA B fragments
__global__ void transpose_v(const float* __restrict__ qkv,
                            _Float16* __restrict__ vT) {
  size_t idx = (size_t)blockIdx.x * blockDim.x + threadIdx.x; // over B*H*64*T
  const size_t n = (size_t)Bc * Hc * HDc * Tc;
  if (idx >= n) return;
  int t = (int)(idx % Tc);
  size_t rest = idx / Tc;
  int d = (int)(rest % HDc); rest /= HDc;
  int h = (int)(rest % Hc);
  int b = (int)(rest / Hc);
  vT[idx] = (_Float16)qkv[(size_t)(b * Tc + t) * QKVN + 2 * Dc + h * HDc + d];
}

// ---------------------------------------------------------------------------
// Kernel 6: flash-attention over splat features. One wave per 16-row q tile.
// logits = qw(16xS) * kw^T (S zero-padded to K=32), online softmax, O += p*V.
__global__ void __launch_bounds__(128)
gsa_attention(const float* __restrict__ P,
              const _Float16* __restrict__ qwh, const _Float16* __restrict__ kwh,
              const _Float16* __restrict__ vT, _Float16* __restrict__ att) {
  __shared__ float pbuf[4][16][36];            // per-wave p-tile transpose pad
  const int lane = threadIdx.x & 31;
  const int wv = threadIdx.x >> 5;
  const int qtg = blockIdx.x * 4 + wv;         // 0 .. B*H*64-1
  const int qt = qtg & 63;
  const int bh = qtg >> 6;
  const int h = bh & 7, b = bh >> 3;
  const float inv_temp = P[8576];
  const int i0 = qt * 16;
  const int tokq0 = b * Tc + i0;
  const int hi = lane >> 4;
  const int m = lane & 15;

  // A_q fragment: qw rows i0..i0+15, K = splat index (S=16, padded to 32)
  v16h aq = {};
  {
    const _Float16* p = qwh + ((size_t)(tokq0 + m) * Hc + h) * Sc;
    if (!hi) {
      #pragma unroll
      for (int t = 0; t < 16; ++t) aq[t] = p[t];   // K = 0..15 (valid)
    }
    // hi lane half covers K=16..31 -> zero pad (already zeroed)
  }

  float mrow[8], lrow[8];
  #pragma unroll
  for (int r = 0; r < 8; ++r) { mrow[r] = -1e30f; lrow[r] = 0.0f; }
  v8f o[4] = {};

  for (int jt = 0; jt < Tc; jt += 32) {
    // ---- logits for two 16-wide kv blocks via WMMA (K = S padded) ----
    v16h bk0 = {}, bk1 = {};
    if (!hi) {  // lanes 0-15 carry K=0..15 (=S); hi half is zero pad
      const _Float16* p0 = kwh + ((size_t)(b * Tc + jt + m) * Hc + h) * Sc;
      const _Float16* p1 = kwh + ((size_t)(b * Tc + jt + 16 + m) * Hc + h) * Sc;
      #pragma unroll
      for (int t = 0; t < 16; ++t) { bk0[t] = p0[t]; bk1[t] = p1[t]; }
    }
    v8f s0 = {}, s1 = {};
    s0 = wmma16(aq, bk0, s0);
    s1 = wmma16(aq, bk1, s1);
    #pragma unroll
    for (int r = 0; r < 8; ++r) { s0[r] *= inv_temp; s1[r] *= inv_temp; }

    // ---- online softmax: row stats live across the 16 lanes of each half ----
    float nm[8], fac[8];
    #pragma unroll
    for (int r = 0; r < 8; ++r) {
      float cm = fmaxf(s0[r], s1[r]);
      cm = fmaxf(cm, __shfl_xor(cm, 1, 32));
      cm = fmaxf(cm, __shfl_xor(cm, 2, 32));
      cm = fmaxf(cm, __shfl_xor(cm, 4, 32));
      cm = fmaxf(cm, __shfl_xor(cm, 8, 32));
      nm[r] = fmaxf(mrow[r], cm);
    }
    v8f p0v, p1v;
    #pragma unroll
    for (int r = 0; r < 8; ++r) {
      p0v[r] = __expf(s0[r] - nm[r]);
      p1v[r] = __expf(s1[r] - nm[r]);
      float sum = p0v[r] + p1v[r];
      sum += __shfl_xor(sum, 1, 32);
      sum += __shfl_xor(sum, 2, 32);
      sum += __shfl_xor(sum, 4, 32);
      sum += __shfl_xor(sum, 8, 32);
      fac[r] = __expf(mrow[r] - nm[r]);
      lrow[r] = lrow[r] * fac[r] + sum;
      mrow[r] = nm[r];
    }
    #pragma unroll
    for (int g = 0; g < 4; ++g)
      #pragma unroll
      for (int r = 0; r < 8; ++r) o[g][r] *= fac[r];

    // ---- C-layout -> A-layout transpose of p through per-wave LDS ----
    #pragma unroll
    for (int r = 0; r < 8; ++r) {
      const int mm = r + hi * 8;
      pbuf[wv][mm][m] = p0v[r];
      pbuf[wv][mm][16 + m] = p1v[r];
    }
    asm volatile("s_wait_dscnt 0x0" ::: "memory"); // wave-local DS ordering
    v16h ap;
    {
      const int kb = hi ? 16 : 0;
      #pragma unroll
      for (int t = 0; t < 16; ++t) ap[t] = (_Float16)pbuf[wv][m][kb + t];
    }

    // ---- O += p (16x32) * V (32x64) via 4 WMMAs ----
    const int kb2 = hi ? 16 : 0;
    #pragma unroll
    for (int g = 0; g < 4; ++g) {
      v16h bv;
      const _Float16* vp =
          vT + ((size_t)(b * Hc + h) * HDc + g * 16 + m) * Tc + jt + kb2;
      #pragma unroll
      for (int t = 0; t < 16; ++t) bv[t] = vp[t];
      o[g] = wmma16(ap, bv, o[g]);
    }
  }

  // normalize and store att[tok][h*64+d] as f16
  float invl[8];
  #pragma unroll
  for (int r = 0; r < 8; ++r) invl[r] = 1.0f / lrow[r];
  #pragma unroll
  for (int g = 0; g < 4; ++g)
    #pragma unroll
    for (int r = 0; r < 8; ++r)
      att[(size_t)(tokq0 + r + hi * 8) * Dc + h * HDc + g * 16 + m] =
          (_Float16)(o[g][r] * invl[r]);
}

// ---------------------------------------------------------------------------
extern "C" void kernel_launch(void* const* d_in, const int* in_sizes, int n_in,
                              void* d_out, int out_size, void* d_ws, size_t ws_size,
                              hipStream_t stream) {
  const float* x      = (const float*)d_in[0];
  const float* bcent  = (const float*)d_in[1];
  const float* cdelt  = (const float*)d_in[2];
  const float* lsc    = (const float*)d_in[3];
  const float* lamp   = (const float*)d_in[4];
  const float* msp    = (const float*)d_in[5];
  const float* temp   = (const float*)d_in[6];
  const float* qkv_w  = (const float*)d_in[7];
  const float* out_w  = (const float*)d_in[8];
  float* out = (float*)d_out;

  char* ws = (char*)d_ws;
  size_t off = 0;
  auto take = [&](size_t bytes) { size_t o = off; off = (off + bytes + 255) & ~(size_t)255; return o; };
  float*    qkv_f32 = (float*)   (ws + take((size_t)MTOK * QKVN * 4));
  _Float16* x_h     = (_Float16*)(ws + take((size_t)MTOK * Dc * 2));
  _Float16* wqkv_h  = (_Float16*)(ws + take((size_t)QKVN * Dc * 2));
  _Float16* wout_h  = (_Float16*)(ws + take((size_t)Dc * Dc * 2));
  _Float16* qw_h    = (_Float16*)(ws + take((size_t)MTOK * 128 * 2));
  _Float16* kw_h    = (_Float16*)(ws + take((size_t)MTOK * 128 * 2));
  _Float16* vT_h    = (_Float16*)(ws + take((size_t)Bc * Hc * HDc * Tc * 2));
  _Float16* att_h   = (_Float16*)(ws + take((size_t)MTOK * Dc * 2));
  float*    P       = (float*)   (ws + take((size_t)(8192 + 3 * 128 + 64) * 4));
  (void)ws_size; (void)in_sizes; (void)n_in; (void)out_size;

  // 1. splat params
  prep_params<<<1, 128, 0, stream>>>(bcent, cdelt, lsc, lamp, msp, temp, P);

  // 2. f16 conversions
  to_f16<<<2048, 256, 0, stream>>>(x, x_h, (size_t)MTOK * Dc);
  to_f16<<<1024, 256, 0, stream>>>(qkv_w, wqkv_h, (size_t)QKVN * Dc);
  to_f16<<<512, 256, 0, stream>>>(out_w, wout_h, (size_t)Dc * Dc);

  // 3. QKV GEMM: (4096x512) x (1536x512)^T -> (4096x1536) f32
  {
    int blocks = (MTOK / 128) * (QKVN / 64);     // 32 * 24 = 768
    gemm_f16_wmma<<<blocks, 256, 0, stream>>>(x_h, wqkv_h, qkv_f32,
                                              MTOK, QKVN, Dc);
  }

  // 4. splat weights qw,kw (f16) + 5. V transpose (f16)
  splat_weights<<<MTOK, 128, 0, stream>>>(qkv_f32, P, qw_h, kw_h);
  {
    size_t n = (size_t)Bc * Hc * HDc * Tc;
    transpose_v<<<(unsigned)((n + 255) / 256), 256, 0, stream>>>(qkv_f32, vT_h);
  }

  // 6. attention: 2048 q-tiles, 4 waves/block
  gsa_attention<<<512, 128, 0, stream>>>(P, qw_h, kw_h, vT_h, att_h);

  // 7. out projection: (4096x512) x (512x512)^T -> d_out f32
  {
    int blocks = (MTOK / 128) * (Dc / 64);       // 32 * 8 = 256
    gemm_f16_wmma<<<blocks, 256, 0, stream>>>(att_h, wout_h, out,
                                              MTOK, Dc, Dc);
  }
}
